// EncodeTransformDecode_55319178772833
// MI455X (gfx1250) — matrix operand order repaired and checked by reading, not compiled, not run on hardware
//
#include <hip/hip_runtime.h>
#include <hip/hip_bf16.h>

// ---------------------------------------------------------------------------
// EncodeTransformDecode graph-transformer for MI455X (gfx1250).
// Dense GEMMs use v_wmma_f32_16x16x32_bf16 (fp32 accumulate), QKV fused into
// one NCOLS=384 GEMM; sparse attention uses f32 VALU + L2 float atomics.
// ---------------------------------------------------------------------------

#define GN 50000
#define GE 600000
#define GD 128
#define GH 4
#define GDH 32
#define GL 4
#define GDFF 512
#define GDIN 16
#define GDOUT 3
#define GQKV 384   // packed q|k|v row stride

typedef __attribute__((ext_vector_type(16))) __bf16 v16bf;
typedef __attribute__((ext_vector_type(8)))  float  v8f;

// Native f32 -> bf16 conversion (lowers to v_cvt_pk_bf16_f32 on gfx1250).
__device__ __forceinline__ __bf16 f32_to_bf16(float f) { return (__bf16)f; }

// --------------------------- WMMA tile loaders -----------------------------
// A (M=16 x K=32 bf16), per ISA 7.12.2: lane l holds row m=l&15; half=l>>4;
// vector elem j -> K = k0 + (j<8 ? half*8+j : 16+half*8+(j-8)).
template <int K>
__device__ __forceinline__ v16bf load_a_f32(const float* __restrict__ A,
                                            int row0, int k0, int lane) {
    const int m = row0 + (lane & 15);
    const int half = lane >> 4;
    const float* Ar = A + (size_t)m * K;
    v16bf a;
#pragma unroll
    for (int j = 0; j < 16; ++j) {
        const int kk = k0 + ((j < 8) ? (half * 8 + j) : (16 + half * 8 + (j - 8)));
        float f = ((K % 32 == 0) || (kk < K)) ? Ar[kk] : 0.0f;
        a[j] = f32_to_bf16(f);
    }
    return a;
}

// B (K=32 x N=16 bf16): lane l holds column n=l&15; half=l>>4;
// elem j -> K = k0 + (j<8 ? half*8+j : 16+half*8+(j-8)).  Bt is pre-transposed [Ncols, K] bf16 so the
// 16 elems per lane are two contiguous 8-elem (16B) runs.
template <int K>
__device__ __forceinline__ v16bf load_b_bf16(const __bf16* __restrict__ Bt,
                                             int col0, int k0, int lane) {
    const int n = col0 + (lane & 15);
    const int half = lane >> 4;
    const __bf16* Br = Bt + (size_t)n * K;
    v16bf b;
#pragma unroll
    for (int j = 0; j < 16; ++j) {
        const int kk = k0 + ((j < 8) ? (half * 8 + j) : (16 + half * 8 + (j - 8)));
        b[j] = ((K % 32 == 0) || (kk < K)) ? Br[kk] : f32_to_bf16(0.0f);
    }
    return b;
}

// --------------------------- generic WMMA GEMM -----------------------------
// C[M,NCOLS] = act( A[M,K] @ B[K,NCOLS] + bias ) (+resid).  One block = 16
// rows, 4 waves, each wave walks column tiles.  M must be a multiple of 16.
template <int K, int NCOLS, bool RELU, bool RESID, bool BIAS>
__global__ void __launch_bounds__(128)
gemm_wmma_kernel(const float* __restrict__ A, const __bf16* __restrict__ Bt,
                 const float* __restrict__ bias, const float* __restrict__ resid,
                 float* __restrict__ C) {
    const int lane = threadIdx.x & 31;
    const int wave = threadIdx.x >> 5;
    const int row0 = blockIdx.x * 16;
    constexpr int KT = (K + 31) / 32;
    constexpr int NT = NCOLS / 16;

    v16bf areg[KT];
#pragma unroll
    for (int kt = 0; kt < KT; ++kt) areg[kt] = load_a_f32<K>(A, row0, kt * 32, lane);

    for (int t = wave; t < NT; t += 4) {
        // branch-free prefetch of the next B tile (global_prefetch_b8)
        const int pt = (t + 4 < NT) ? (t + 4) : t;
        __builtin_prefetch(Bt + (size_t)pt * 16 * K + (lane << 6), 0, 1);
        // batch all B-tile loads first so they issue as one clause,
        // then drain them through the WMMA chain.
        v16bf breg[KT];
#pragma unroll
        for (int kt = 0; kt < KT; ++kt) breg[kt] = load_b_bf16<K>(Bt, t * 16, kt * 32, lane);
        v8f c = {};
#pragma unroll
        for (int kt = 0; kt < KT; ++kt)
            c = __builtin_amdgcn_wmma_f32_16x16x32_bf16(false, areg[kt], false, breg[kt],
                                                        (short)0, c, false, false);
        const int n = t * 16 + (lane & 15);
        const int mb = row0 + (lane >> 4) * 8;
        float bv = BIAS ? bias[n] : 0.0f;
#pragma unroll
        for (int r = 0; r < 8; ++r) {
            float val = c[r] + bv;
            if (RELU) val = fmaxf(val, 0.0f);
            size_t off = (size_t)(mb + r) * NCOLS + n;
            if (RESID) val += resid[off];
            C[off] = val;
        }
    }
}

// --------------------------- fused FFN block -------------------------------
// x += relu(h @ W1 + b1) @ W2 + b2.   mid tile (16x512) staged in LDS as bf16.
__global__ void __launch_bounds__(128)
ffn_wmma_kernel(const float* __restrict__ h,
                const __bf16* __restrict__ w1t, const float* __restrict__ b1,
                const __bf16* __restrict__ w2t, const float* __restrict__ b2,
                float* __restrict__ x) {
    __shared__ __bf16 mid[16][GDFF + 16];
    const int lane = threadIdx.x & 31;
    const int wave = threadIdx.x >> 5;
    const int row0 = blockIdx.x * 16;

    // stage 1: mid = relu(h @ W1 + b1) -> LDS (bf16)
    v16bf areg[4];
#pragma unroll
    for (int kt = 0; kt < 4; ++kt) areg[kt] = load_a_f32<GD>(h, row0, kt * 32, lane);

    for (int t = wave; t < GDFF / 16; t += 4) {
        const int pt = (t + 4 < GDFF / 16) ? (t + 4) : t;
        __builtin_prefetch(w1t + (size_t)pt * 16 * GD + (lane << 6), 0, 1);
        v16bf breg[4];
#pragma unroll
        for (int kt = 0; kt < 4; ++kt) breg[kt] = load_b_bf16<GD>(w1t, t * 16, kt * 32, lane);
        v8f c = {};
#pragma unroll
        for (int kt = 0; kt < 4; ++kt)
            c = __builtin_amdgcn_wmma_f32_16x16x32_bf16(false, areg[kt], false, breg[kt],
                                                        (short)0, c, false, false);
        const int n = t * 16 + (lane & 15);
        const int mb = (lane >> 4) * 8;
#pragma unroll
        for (int r = 0; r < 8; ++r)
            mid[mb + r][n] = f32_to_bf16(fmaxf(c[r] + b1[n], 0.0f));
    }
    __syncthreads();

    // stage 2: x += mid @ W2 + b2 (K=512: partial unroll so loads batch ahead)
    for (int t = wave; t < GD / 16; t += 4) {
        v8f c = {};
#pragma unroll 4
        for (int kt = 0; kt < GDFF / 32; ++kt) {
            v16bf a;
            const int m = lane & 15;
            const int half = lane >> 4;
#pragma unroll
            for (int j = 0; j < 16; ++j) {
                int kk = kt * 32 + ((j < 8) ? (half * 8 + j) : (16 + half * 8 + (j - 8)));
                a[j] = mid[m][kk];
            }
            v16bf b = load_b_bf16<GDFF>(w2t, t * 16, kt * 32, lane);
            c = __builtin_amdgcn_wmma_f32_16x16x32_bf16(false, a, false, b,
                                                        (short)0, c, false, false);
        }
        const int n = t * 16 + (lane & 15);
        const int mb = row0 + (lane >> 4) * 8;
#pragma unroll
        for (int r = 0; r < 8; ++r) {
            size_t off = (size_t)(mb + r) * GD + n;
            x[off] = c[r] + b2[n] + x[off];
        }
    }
}

// --------------------------- LayerNorm (one wave / node) -------------------
__global__ void __launch_bounds__(128)
ln_kernel(const float* __restrict__ x, const float* __restrict__ s,
          const float* __restrict__ b, float* __restrict__ o, int nnodes) {
    const int lane = threadIdx.x & 31;
    const int wave = threadIdx.x >> 5;
    const int node = blockIdx.x * 4 + wave;
    if (node >= nnodes) return;
    const float4 v = ((const float4*)(x + (size_t)node * GD))[lane];
    float sum = v.x + v.y + v.z + v.w;
    float sq  = v.x * v.x + v.y * v.y + v.z * v.z + v.w * v.w;
#pragma unroll
    for (int d = 16; d >= 1; d >>= 1) {
        sum += __shfl_xor(sum, d, 32);
        sq  += __shfl_xor(sq,  d, 32);
    }
    const float mu = sum * (1.0f / GD);
    const float var = sq * (1.0f / GD) - mu * mu;
    const float rs = rsqrtf(var + 1e-5f);
    const float4 sv = ((const float4*)s)[lane];
    const float4 bv = ((const float4*)b)[lane];
    float4 ov;
    ov.x = (v.x - mu) * rs * sv.x + bv.x;
    ov.y = (v.y - mu) * rs * sv.y + bv.y;
    ov.z = (v.z - mu) * rs * sv.z + bv.z;
    ov.w = (v.w - mu) * rs * sv.w + bv.w;
    ((float4*)(o + (size_t)node * GD))[lane] = ov;
}

// --------------------------- sparse attention pieces -----------------------
__device__ __forceinline__ unsigned f32_sortable(float f) {
    unsigned u = __builtin_bit_cast(unsigned, f);
    return (u & 0x80000000u) ? ~u : (u | 0x80000000u);
}

__global__ void attn_init_kernel(unsigned* __restrict__ menc, int n) {
    int i = blockIdx.x * blockDim.x + threadIdx.x;
    if (i < n) menc[i] = 0x007FFFFFu;  // sortable(-inf)
}

// qkv packed [N, 384]: q = cols 0..127, k = 128..255, v = 256..383
__global__ void edge_score_kernel(const float* __restrict__ qkv,
                                  const int* __restrict__ snd, const int* __restrict__ rcv,
                                  float* __restrict__ scores, unsigned* __restrict__ menc,
                                  int e_total) {
    int t = blockIdx.x * blockDim.x + threadIdx.x;
    if (t >= e_total) return;
    const int e = t >> 2, hh = t & 3;
    const int s = snd[e], r = rcv[e];
    const float* qp = qkv + (size_t)r * GQKV + hh * GDH;
    const float* kp = qkv + (size_t)s * GQKV + GD + hh * GDH;
    float acc = 0.0f;
#pragma unroll
    for (int d = 0; d < GDH; ++d) acc += qp[d] * kp[d];
    acc *= 0.17677669529663687f;  // 1/sqrt(32)
    scores[t] = acc;
    atomicMax(&menc[r * 4 + hh], f32_sortable(acc));
}

__global__ void seg_finalize_kernel(const unsigned* __restrict__ menc,
                                    float* __restrict__ m, float* __restrict__ denom,
                                    float* __restrict__ msg, int n) {
    int i = blockIdx.x * blockDim.x + threadIdx.x;
    if (i >= n) return;
    unsigned su = menc[i];
    unsigned u = (su & 0x80000000u) ? (su ^ 0x80000000u) : ~su;
    float f = __builtin_bit_cast(float, u);
    if (!(fabsf(f) < 3.0e38f)) f = 0.0f;  // where(isfinite, m, 0)
    m[i] = f;
    denom[i] = 0.0f;
    float* mp = msg + (size_t)(i >> 2) * GD + (i & 3) * GDH;
#pragma unroll
    for (int d = 0; d < GDH; ++d) mp[d] = 0.0f;
}

__global__ void edge_accum_kernel(const float* __restrict__ scores, const float* __restrict__ m,
                                  const float* __restrict__ qkv, const int* __restrict__ snd,
                                  const int* __restrict__ rcv, float* __restrict__ denom,
                                  float* __restrict__ msg, int e_total) {
    int t = blockIdx.x * blockDim.x + threadIdx.x;
    if (t >= e_total) return;
    const int e = t >> 2, hh = t & 3;
    const int s = snd[e], r = rcv[e];
    const float a = __expf(scores[t] - m[r * 4 + hh]);
    atomicAdd(&denom[r * 4 + hh], a);
    const float* vp = qkv + (size_t)s * GQKV + 2 * GD + hh * GDH;
    float* mp = msg + (size_t)r * GD + hh * GDH;
#pragma unroll
    for (int d = 0; d < GDH; ++d) atomicAdd(&mp[d], a * vp[d]);
}

__global__ void attn_div_kernel(float* __restrict__ msg, const float* __restrict__ denom,
                                int total) {
    int i = blockIdx.x * blockDim.x + threadIdx.x;
    if (i >= total) return;
    const int node = i >> 7;
    const int hh = (i >> 5) & 3;
    msg[i] = msg[i] / (denom[node * 4 + hh] + 1e-9f);
}

// --------------------------- decoder ---------------------------------------
__global__ void decode_kernel(const float* __restrict__ x, const float* __restrict__ w1,
                              const float* __restrict__ b1, const float* __restrict__ w2,
                              const float* __restrict__ b2, float* __restrict__ out, int n) {
    int node = blockIdx.x * blockDim.x + threadIdx.x;
    if (node >= n) return;
    const float* xr = x + (size_t)node * GD;
    float a0 = b1[0], a1 = b1[1], a2 = b1[2];
    for (int i = 0; i < GD; ++i) {
        const float xv = xr[i];
        a0 += xv * w1[i * 3 + 0];
        a1 += xv * w1[i * 3 + 1];
        a2 += xv * w1[i * 3 + 2];
    }
    a0 = fmaxf(a0, 0.0f); a1 = fmaxf(a1, 0.0f); a2 = fmaxf(a2, 0.0f);
#pragma unroll
    for (int j = 0; j < 3; ++j)
        out[(size_t)node * 3 + j] = b2[j] + a0 * w2[0 * 3 + j] + a1 * w2[1 * 3 + j] + a2 * w2[2 * 3 + j];
}

// --------------------------- weight prep: transpose + bf16 -----------------
__global__ void prep_weight_kernel(const float* __restrict__ W, __bf16* __restrict__ out,
                                   int K, int Nc) {
    int i = blockIdx.x * blockDim.x + threadIdx.x;
    if (i >= K * Nc) return;
    const int k = i / Nc, n = i % Nc;
    out[(size_t)n * K + k] = f32_to_bf16(W[i]);
}

// ---------------------------------------------------------------------------
extern "C" void kernel_launch(void* const* d_in, const int* in_sizes, int n_in,
                              void* d_out, int out_size, void* d_ws, size_t ws_size,
                              hipStream_t stream) {
    const float* features = (const float*)d_in[0];
    const int* senders    = (const int*)d_in[1];
    const int* receivers  = (const int*)d_in[2];
    const float* enc_w1 = (const float*)d_in[3];
    const float* enc_b1 = (const float*)d_in[4];
    const float* enc_w2 = (const float*)d_in[5];
    const float* enc_b2 = (const float*)d_in[6];
    const float* wq = (const float*)d_in[7];
    const float* wk = (const float*)d_in[8];
    const float* wv = (const float*)d_in[9];
    const float* wo = (const float*)d_in[10];
    const float* ln1_s = (const float*)d_in[11];
    const float* ln1_b = (const float*)d_in[12];
    const float* ffn_w1 = (const float*)d_in[13];
    const float* ffn_b1 = (const float*)d_in[14];
    const float* ffn_w2 = (const float*)d_in[15];
    const float* ffn_b2 = (const float*)d_in[16];
    const float* ln2_s = (const float*)d_in[17];
    const float* ln2_b = (const float*)d_in[18];
    const float* dec_w1 = (const float*)d_in[19];
    const float* dec_b1 = (const float*)d_in[20];
    const float* dec_w2 = (const float*)d_in[21];
    const float* dec_b2 = (const float*)d_in[22];

    // ---- workspace carve-up (all 256B aligned) ----
    char* ws = (char*)d_ws;
    size_t off = 0;
    auto carve = [&](size_t bytes) {
        void* p = ws + off;
        off += (bytes + 255) & ~(size_t)255;
        return p;
    };
    float* x     = (float*)carve((size_t)GN * GD * 4);
    float* h     = (float*)carve((size_t)GN * GD * 4);
    float* qkv   = (float*)carve((size_t)GN * GQKV * 4);
    float* msg   = (float*)carve((size_t)GN * GD * 4);
    float* scores = (float*)carve((size_t)GE * GH * 4);
    float* m      = (float*)carve((size_t)GN * GH * 4);
    float* denom  = (float*)carve((size_t)GN * GH * 4);
    unsigned* menc = (unsigned*)carve((size_t)GN * GH * 4);
    __bf16* enc1t = (__bf16*)carve((size_t)GD * GDIN * 2);
    __bf16* enc2t = (__bf16*)carve((size_t)GD * GD * 2);
    __bf16* qkvt  = (__bf16*)carve((size_t)GL * GQKV * GD * 2);  // [384,128] per layer
    __bf16* wot   = (__bf16*)carve((size_t)GL * GD * GD * 2);
    __bf16* f1t   = (__bf16*)carve((size_t)GL * GD * GDFF * 2);
    __bf16* f2t   = (__bf16*)carve((size_t)GL * GDFF * GD * 2);

    auto prep = [&](const float* W, __bf16* o, int K, int Nc) {
        int total = K * Nc;
        prep_weight_kernel<<<(total + 255) / 256, 256, 0, stream>>>(W, o, K, Nc);
    };

    // ---- weight prep (bf16, pre-transposed to [out, in]) ----
    prep(enc_w1, enc1t, GDIN, GD);
    prep(enc_w2, enc2t, GD, GD);
    for (int l = 0; l < GL; ++l) {
        __bf16* qt = qkvt + (size_t)l * GQKV * GD;
        prep(wq + (size_t)l * GD * GD, qt + 0 * GD * GD, GD, GD);   // rows   0..127
        prep(wk + (size_t)l * GD * GD, qt + 1 * GD * GD, GD, GD);   // rows 128..255
        prep(wv + (size_t)l * GD * GD, qt + 2 * GD * GD, GD, GD);   // rows 256..383
        prep(wo + (size_t)l * GD * GD, wot + (size_t)l * GD * GD, GD, GD);
        prep(ffn_w1 + (size_t)l * GD * GDFF, f1t + (size_t)l * GDFF * GD, GD, GDFF);
        prep(ffn_w2 + (size_t)l * GDFF * GD, f2t + (size_t)l * GD * GDFF, GDFF, GD);
    }

    const int gemm_blocks = GN / 16;  // 3125
    const int nh = GN * GH;
    const int eh = GE * GH;

    // ---- encoder: x = relu(features@W1+b1) @ W2 + b2 ----
    gemm_wmma_kernel<GDIN, GD, true, false, true>
        <<<gemm_blocks, 128, 0, stream>>>(features, enc1t, enc_b1, nullptr, h);
    gemm_wmma_kernel<GD, GD, false, false, true>
        <<<gemm_blocks, 128, 0, stream>>>(h, enc2t, enc_b2, nullptr, x);

    // ---- L transformer blocks ----
    for (int l = 0; l < GL; ++l) {
        const size_t fl = (size_t)l * GD * GDFF;
        // h = LN1(x)
        ln_kernel<<<GN / 4, 128, 0, stream>>>(x, ln1_s + l * GD, ln1_b + l * GD, h, GN);
        // qkv = h @ [Wq|Wk|Wv]  (fused, A-tile reused across 24 column tiles)
        gemm_wmma_kernel<GD, GQKV, false, false, false>
            <<<gemm_blocks, 128, 0, stream>>>(h, qkvt + (size_t)l * GQKV * GD,
                                              nullptr, nullptr, qkv);
        // sparse edge softmax + aggregation
        attn_init_kernel<<<(nh + 255) / 256, 256, 0, stream>>>(menc, nh);
        edge_score_kernel<<<(eh + 255) / 256, 256, 0, stream>>>(qkv, senders, receivers,
                                                                scores, menc, eh);
        seg_finalize_kernel<<<(nh + 255) / 256, 256, 0, stream>>>(menc, m, denom, msg, nh);
        edge_accum_kernel<<<(eh + 255) / 256, 256, 0, stream>>>(scores, m, qkv, senders,
                                                                receivers, denom, msg, eh);
        attn_div_kernel<<<(GN * GD + 255) / 256, 256, 0, stream>>>(msg, denom, GN * GD);
        // x += msg @ Wo
        gemm_wmma_kernel<GD, GD, false, true, false>
            <<<gemm_blocks, 128, 0, stream>>>(msg, wot + (size_t)l * GD * GD, nullptr, x, x);
        // h = LN2(x); x += relu(h@W1+b1)@W2+b2 (fused)
        ln_kernel<<<GN / 4, 128, 0, stream>>>(x, ln2_s + l * GD, ln2_b + l * GD, h, GN);
        ffn_wmma_kernel<<<gemm_blocks, 128, 0, stream>>>(h, f1t + fl, ffn_b1 + l * GDFF,
                                                         f2t + fl, ffn_b2 + l * GD, x);
    }

    // ---- decoder ----
    decode_kernel<<<(GN + 255) / 256, 256, 0, stream>>>(x, dec_w1, dec_b1, dec_w2, dec_b2,
                                                        (float*)d_out, GN);
}
